// HartleySpectralConv2d_6064493822414
// MI455X (gfx1250) — compile-verified, matrix operand order checked by reading, not run on hardware
//
#include <hip/hip_runtime.h>

typedef _Float16 v16h __attribute__((ext_vector_type(16)));
typedef _Float16 v8h  __attribute__((ext_vector_type(8)));
typedef _Float16 v4h  __attribute__((ext_vector_type(4)));
typedef float    v8f  __attribute__((ext_vector_type(8)));

#define PI_F 3.14159265358979323846f

// ---- workspace layout (bytes) ----
#define OFF_BX   0u                                   // bxT [128 n][256 x]   f16  64 KB
#define OFF_AY   (OFF_BX   + 128u*256u*2u)            // ay  [128 m][512 k]   f16 128 KB
#define OFF_BC   (OFF_AY   + 128u*512u*2u)            // bcT [512 n][64 fx]   f16  64 KB
#define OFF_AD   (OFF_BC   + 512u*64u*2u)             // ad  [256 y][128 k]   f16  64 KB
#define OFF_WMIX (OFF_AD   + 256u*128u*2u)            // [4096 mode][64 co][128 k] 64 MB
#define OFF_BEO  (OFF_WMIX + 4096u*64u*128u*2u)       // [2048 b*128+k][4096 mode] 16 MB
#define OFF_AEOT (OFF_BEO  + 2048u*4096u*2u)          // [4096 mode][16 b][128 k]  16 MB
#define OFF_TOTN (OFF_AEOT + 4096u*2048u*2u)          // [4096 mode][1024 b*co]     8 MB
#define OFF_TOTT (OFF_TOTN + 4096u*1024u*2u)          // [1024 b*co][4096 mode]     8 MB
// total ~112.3 MB

// ---------- WMMA fragment helper ----------
// Works for BOTH operands when data is K-contiguous per lane:
//   A: linep = row pointer (row = m0+r), 16-bit A 16x32 layout (ISA 7.12.2)
//   B: linep = column pointer (col = n0+r), K down the line (N-major storage)
// Lane needs halves at k0+half*8+[0,8) and k0+16+half*8+[0,8): two 16B loads.
__device__ __forceinline__ v16h load_frag(const _Float16* linep, int k0, int half) {
  const _Float16* p = linep + k0 + (half << 3);
  v8h lo = *(const v8h*)(p);
  v8h hi = *(const v8h*)(p + 16);
  return __builtin_shufflevector(lo, hi, 0, 1, 2, 3, 4, 5, 6, 7,
                                 8, 9, 10, 11, 12, 13, 14, 15);
}

__device__ __forceinline__ v8f wmma_f16(v16h a, v16h b, v8f c) {
  return __builtin_amdgcn_wmma_f32_16x16x32_f16(false, a, false, b, (short)0, c,
                                                false, false);
}

__device__ __forceinline__ v8h cvt8(v8f a) {
  v8h h;
#pragma unroll
  for (int j = 0; j < 8; ++j) h[j] = (_Float16)a[j];
  return h;
}

// exact phase: (2*pi/256) * ((s*n) mod 256)
__device__ __forceinline__ float casarg(int s, int n) {
  int ph = (s * n) & 255;
  return (2.0f * PI_F / 256.0f) * (float)ph;
}

// ---------- basis tables (B-side tables stored K-contiguous) ----------
__global__ __launch_bounds__(256) void k_tables(_Float16* bxT, _Float16* ay,
                                                _Float16* bcT, _Float16* ad) {
  int idx = blockIdx.x * 256 + threadIdx.x;
  if (idx < 32768) {                 // bxT[n][x]: n<64 cos, n>=64 -sin (x-dim DFT)
    int n = idx >> 8, x = idx & 255;
    float v = (n < 64) ? cosf(casarg(n - 32, x)) : -sinf(casarg(n - 96, x));
    bxT[idx] = (_Float16)v;
  } else if (idx < 98304) {          // ay[m][k]: even rows [cos|sin], odd [sin|-cos]
    int i = idx - 32768;
    int m = i >> 9, k = i & 511;
    int s = (m & 63) - 32, y = k & 255;
    float th = casarg(s, y), v;
    if (m < 64) v = (k < 256) ? cosf(th) : sinf(th);
    else        v = (k < 256) ? sinf(th) : -cosf(th);
    ay[i] = (_Float16)v;
  } else if (idx < 131072) {         // bcT[n][fx]: n<256 cos, n>=256 -sin
    int i = idx - 98304;
    int n = i >> 6, fx = i & 63;
    float th = casarg(fx - 32, n & 255);
    bcT[i] = (_Float16)((n < 256) ? cosf(th) : -sinf(th));
  } else if (idx < 163840) {         // ad[y][k]: k<64 (cos+sin), k>=64 (sin-cos)
    int i = idx - 131072;
    int y = i >> 7, k = i & 127;
    float th = casarg((k & 63) - 32, y);
    float c = cosf(th), sn = sinf(th);
    ad[i] = (_Float16)((k < 64) ? (c + sn) : (sn - c));
  }
}

// wmix[mode][co][k]: k<64 -> w[ci=k][co][mode]; k>=64 -> w[ci][co][flip(mode)]
// LDS-tiled so reads AND both writes are coalesced.
__global__ __launch_bounds__(256) void k_wmix(const float* __restrict__ w,
                                              _Float16* __restrict__ wmix) {
  __shared__ _Float16 sW[64][65];              // [mx][ci], padded
  const int co = blockIdx.x >> 6, my = blockIdx.x & 63;
  for (int i = threadIdx.x; i < 4096; i += 256) {
    int ci = i >> 6, mx = i & 63;              // consecutive mx -> coalesced read
    sW[mx][ci] = (_Float16)w[(size_t)((ci << 6) | co) * 4096 + (my << 6) + mx];
  }
  __syncthreads();
  for (int i = threadIdx.x; i < 4096; i += 256) {
    int mx = i >> 6, ci = i & 63;              // consecutive ci -> coalesced writes
    _Float16 v = sW[mx][ci];
    wmix[(size_t)((my << 6) | mx) * 8192 + (co << 7) + ci] = v;
    int fmy = (64 - my) & 63, fmx = (64 - mx) & 63;
    wmix[(size_t)((fmy << 6) | fmx) * 8192 + (co << 7) + 64 + ci] = v;
  }
}

// generic 64x64-tiled f16 transpose: dst[C][R] = src[R][C]  (R,C multiples of 64)
__global__ __launch_bounds__(256) void k_transpose(const _Float16* __restrict__ src,
                                                   _Float16* __restrict__ dst,
                                                   int R, int C) {
  __shared__ _Float16 tile[64][65];
  int tilesC = C >> 6;
  int r0 = (blockIdx.x / tilesC) << 6, c0 = (blockIdx.x % tilesC) << 6;
  for (int i = threadIdx.x; i < 4096; i += 256) {
    int rr = i >> 6, cc = i & 63;
    tile[rr][cc] = src[(size_t)(r0 + rr) * C + c0 + cc];
  }
  __syncthreads();
  for (int i = threadIdx.x; i < 4096; i += 256) {
    int rr = i >> 6, cc = i & 63;
    dst[(size_t)(c0 + rr) * R + r0 + cc] = tile[cc][rr];
  }
}

// ---------- forward: partial DFT (stages A+B fused per image) ----------
__global__ __launch_bounds__(256) void k_forward(const float* __restrict__ x,
                                                 const _Float16* __restrict__ bxT,
                                                 const _Float16* __restrict__ ay,
                                                 _Float16* __restrict__ beo) {
  extern __shared__ char smem[];
  _Float16* sBX = (_Float16*)smem;               // bxT [128 n][256 x]   (64 KB)
  _Float16* sT  = (_Float16*)(smem + 65536);     // T   [128 col][256 y] (64 KB) col-major
  _Float16* sX  = (_Float16*)(smem + 131072);    // x   [16 y][256 x]    ( 8 KB)
  const int tid = threadIdx.x;
  const int lane = tid & 31, wave = tid >> 5;
  const int r = lane & 15, half = lane >> 4;
  const int img = blockIdx.x;                    // b*64 + ci
  const int bidx = img >> 6, ci = img & 63;
  const float* xi = x + (size_t)img * 65536;

  {  // stage basis into LDS (16B copies)
    const uint4* gsrc = (const uint4*)bxT;
    uint4* gdst = (uint4*)sBX;
    for (int i = tid; i < 4096; i += 256) gdst[i] = gsrc[i];
  }
  __syncthreads();

  // Stage A: T[col=128][y] = (x @ [cos|-sin]) transposed   (M=256, N=128, K=256)
  for (int rb = 0; rb < 16; ++rb) {
    const float4* xv = (const float4*)xi + rb * 1024;
    for (int i = tid; i < 1024; i += 256) {
      float4 v = xv[i];
      v4h h = { (_Float16)v.x, (_Float16)v.y, (_Float16)v.z, (_Float16)v.w };
      *(v4h*)(sX + i * 4) = h;
    }
    __syncthreads();
    v8f acc = {};
#pragma unroll
    for (int kc = 0; kc < 8; ++kc) {
      v16h a = load_frag(sX + r * 256, kc * 32, half);            // rows y
      v16h b = load_frag(sBX + (wave * 16 + r) * 256, kc * 32, half); // cols n
      acc = wmma_f16(a, b, acc);
    }
    // D rows are consecutive y -> one 16B LDS store per lane (column-major T)
    *(v8h*)(sT + (wave * 16 + r) * 256 + rb * 16 + half * 8) = cvt8(acc);
    __syncthreads();
  }

  // Stage B: [even;odd][m=128][fx=64] = ay[128x512] @ [T_re;T_im][512x64]
  for (int tt = 0; tt < 4; ++tt) {
    int t = wave * 4 + tt;
    int m0 = (t >> 2) * 16, n0 = (t & 3) * 16;
    v8f acc = {};
#pragma unroll
    for (int kc = 0; kc < 16; ++kc) {
      v16h a = load_frag(ay + (m0 + r) * 512, kc * 32, half);
      // B K-line: re -> col fx, im -> col 64+fx (chunk never crosses k=256)
      const _Float16* bl = (kc < 8) ? (sT + (n0 + r) * 256)
                                    : (sT + (64 + n0 + r) * 256 - 256);
      v16h b = load_frag(bl, kc * 32, half);
      acc = wmma_f16(a, b, acc);
    }
#pragma unroll
    for (int j = 0; j < 8; ++j) {       // natural layout: [b*128 + eo*64+ci][mode]
      int m = m0 + j + half * 8;
      int eo = m >> 6, fy = m & 63, fx = n0 + r;
      beo[(size_t)(bidx * 128 + eo * 64 + ci) * 4096 + (fy << 6) + fx] =
          (_Float16)acc[j];
    }
  }
}

// ---------- per-mode channel mixing: [16b x 128] @ [128 x 64co] ----------
__global__ __launch_bounds__(128) void k_mix(const _Float16* __restrict__ aeoT,
                                             const _Float16* __restrict__ wmix,
                                             _Float16* __restrict__ totN) {
  const int tid = threadIdx.x;
  const int lane = tid & 31, wave = tid >> 5;
  const int r = lane & 15, half = lane >> 4;
  const int mode = blockIdx.x;
  const _Float16* A  = aeoT + (size_t)mode * 2048;   // [b=16][k=128]
  const _Float16* Bm = wmix + (size_t)mode * 8192;   // [co=64][k=128]
  const int n0 = wave * 16;
  v8f acc = {};
#pragma unroll
  for (int kc = 0; kc < 4; ++kc) {
    v16h a = load_frag(A + r * 128, kc * 32, half);          // row b = r
    v16h b = load_frag(Bm + (n0 + r) * 128, kc * 32, half);  // col co
    acc = wmma_f16(a, b, acc);
  }
#pragma unroll
  for (int j = 0; j < 8; ++j) {       // natural layout: [mode][b*64+co]
    totN[(size_t)mode * 1024 + (j + half * 8) * 64 + n0 + r] = (_Float16)acc[j];
  }
}

// ---------- inverse: partial iDHT (stages C+D fused per image) + bias + relu ----------
__global__ __launch_bounds__(256) void k_inverse(const _Float16* __restrict__ totT,
                                                 const _Float16* __restrict__ bcT,
                                                 const _Float16* __restrict__ ad,
                                                 const float* __restrict__ bias,
                                                 float* __restrict__ out) {
  extern __shared__ char smem[];
  _Float16* sV = (_Float16*)smem;        // V [512 col x'|re,im][64 fy] col-major (64 KB)
  const int tid = threadIdx.x;
  const int lane = tid & 31, wave = tid >> 5;
  const int r = lane & 15, half = lane >> 4;
  const int img = blockIdx.x;            // b*64 + co
  const int co = img & 63;
  const _Float16* Ablk = totT + (size_t)img * 4096;   // [fy][fx]

  // Stage C: V[64 fy x 512] = total[64x64] @ bcT^T, stored column-major
  for (int tt = 0; tt < 16; ++tt) {
    int t = wave * 16 + tt;
    int m0 = (t >> 5) * 16, n0 = (t & 31) * 16;
    v8f acc = {};
#pragma unroll
    for (int kc = 0; kc < 2; ++kc) {
      v16h a = load_frag(Ablk + (m0 + r) * 64, kc * 32, half);
      v16h b = load_frag(bcT + (n0 + r) * 64, kc * 32, half);
      acc = wmma_f16(a, b, acc);
    }
    // D rows consecutive fy -> one 16B LDS store per lane
    *(v8h*)(sV + (n0 + r) * 64 + m0 + half * 8) = cvt8(acc);
  }
  __syncthreads();

  const float bco = bias[co];
  float* oimg = out + (size_t)img * 65536;
  // Stage D: out[256 y][256 x] = ad[256x128] @ [V_re;V_im][128x256]
  for (int tt = 0; tt < 32; ++tt) {
    int t = wave * 32 + tt;
    int m0 = (t >> 4) * 16, n0 = (t & 15) * 16;
    v8f acc = {};
#pragma unroll
    for (int kc = 0; kc < 4; ++kc) {
      v16h a = load_frag(ad + (m0 + r) * 128, kc * 32, half);
      // B K-line: re -> col x, im -> col 256+x (chunk never crosses k=64)
      const _Float16* bl = (kc < 2) ? (sV + (n0 + r) * 64)
                                    : (sV + (256 + n0 + r) * 64 - 64);
      v16h b = load_frag(bl, kc * 32, half);
      acc = wmma_f16(a, b, acc);
    }
#pragma unroll
    for (int j = 0; j < 8; ++j) {
      float v = acc[j] * (1.0f / 65536.0f) + bco;
      v = v > 0.0f ? v : 0.0f;
      oimg[(m0 + j + half * 8) * 256 + n0 + r] = v;   // coalesced f32 stores
    }
  }
}

extern "C" void kernel_launch(void* const* d_in, const int* in_sizes, int n_in,
                              void* d_out, int out_size, void* d_ws, size_t ws_size,
                              hipStream_t stream) {
  const float* x    = (const float*)d_in[0];   // [16,64,256,256]
  const float* w    = (const float*)d_in[1];   // [64,64,64,64]
  const float* bias = (const float*)d_in[2];   // [64,1,1]
  float* out = (float*)d_out;

  char* ws = (char*)d_ws;
  _Float16* bxT  = (_Float16*)(ws + OFF_BX);
  _Float16* ay   = (_Float16*)(ws + OFF_AY);
  _Float16* bcT  = (_Float16*)(ws + OFF_BC);
  _Float16* ad   = (_Float16*)(ws + OFF_AD);
  _Float16* wmix = (_Float16*)(ws + OFF_WMIX);
  _Float16* beo  = (_Float16*)(ws + OFF_BEO);
  _Float16* aeoT = (_Float16*)(ws + OFF_AEOT);
  _Float16* totN = (_Float16*)(ws + OFF_TOTN);
  _Float16* totT = (_Float16*)(ws + OFF_TOTT);

  (void)in_sizes; (void)n_in; (void)out_size; (void)ws_size;

  hipFuncSetAttribute(reinterpret_cast<const void*>(k_forward),
                      hipFuncAttributeMaxDynamicSharedMemorySize, 139264);
  hipFuncSetAttribute(reinterpret_cast<const void*>(k_inverse),
                      hipFuncAttributeMaxDynamicSharedMemorySize, 65536);

  k_tables   <<<640,  256, 0,      stream>>>(bxT, ay, bcT, ad);
  k_wmix     <<<4096, 256, 0,      stream>>>(w, wmix);
  k_forward  <<<1024, 256, 139264, stream>>>(x, bxT, ay, beo);
  k_transpose<<<2048, 256, 0,      stream>>>(beo, aeoT, 2048, 4096);
  k_mix      <<<4096, 128, 0,      stream>>>(aeoT, wmix, totN);
  k_transpose<<<1024, 256, 0,      stream>>>(totN, totT, 4096, 1024);
  k_inverse  <<<1024, 256, 65536,  stream>>>(totT, bcT, ad, bias, out);
}
